// HybridQTransformerTorch_65481071400554
// MI455X (gfx1250) — compile-verified, matrix operand order checked by reading, not run on hardware
//
#include <hip/hip_runtime.h>
#include <hip/hip_bf16.h>

#define DI __device__ __forceinline__

typedef __attribute__((ext_vector_type(16))) __bf16 bv16;
typedef __attribute__((ext_vector_type(8)))  float  v8f;

union FragU { bv16 v; uint4 q[2]; };

// Load one 16x32 bf16 WMMA fragment (A or B^T, row-major, 16B-aligned rows).
// Lane L: row = L%16, K-halves at element offsets half*8 and 16+half*8.
// Caller passes p already offset to (row*ld + half*8).
DI bv16 lds_frag(const unsigned short* p) {
  FragU f;
  f.q[0] = *(const uint4*)(p);
  f.q[1] = *(const uint4*)(p + 16);
  return f.v;
}

DI v8f bmma(bv16 a, bv16 b, v8f c) {
  return __builtin_amdgcn_wmma_f32_16x16x32_bf16(false, a, false, b, (short)0, c,
                                                 false, false);
}

DI unsigned short f2bf(float f) {  // round-to-nearest-even f32 -> bf16
  union { float f; unsigned int u; } x; x.f = f;
  unsigned int r = x.u + 0x7FFFu + ((x.u >> 16) & 1u);
  return (unsigned short)(r >> 16);
}

DI unsigned int pk2(float a, float b) {  // pack 2 bf16 into u32
  return (unsigned int)f2bf(a) | ((unsigned int)f2bf(b) << 16);
}

DI float posenc_val(int s, int n) {
  int i2 = n & ~1;
  float d = __expf((float)i2 * (-9.210340371976184f / 512.0f)); // -ln(1e4)/512
  float a = (float)s * d;
  return (n & 1) ? __cosf(a) : __sinf(a);
}

// ---------------------------------------------------------------------------
// Generic tiled GEMM: C[M,N] = alpha * A[M,K] * Bt[N,K]^T (+bias,+posenc,bf16)
// Block tile 128x128, K chunk 32, double-buffered LDS (1 barrier / K-step).
// ---------------------------------------------------------------------------
template<bool OUT_BF16, bool POSENC>
__global__ __launch_bounds__(256) void gemm_kernel(
    const float* __restrict__ A, int lda,
    const float* __restrict__ Bt, int ldb,
    void* __restrict__ Cptr, int ldc,
    int K, const float* __restrict__ bias, float alpha)
{
  __shared__ __align__(16) unsigned short As[2][128 * 40];
  __shared__ __align__(16) unsigned short Bs[2][128 * 40];
  const int t = threadIdx.x;
  const int m0 = blockIdx.y * 128, n0 = blockIdx.x * 128;
  const int lane = t & 31, r16 = lane & 15, half = lane >> 4;
  const int wv = t >> 5, waveM = wv >> 1, waveN = wv & 1;
  const int srow = t >> 3, scol = (t & 7) * 4;

  auto stage = [&](int kc, int buf) {
    #pragma unroll
    for (int p = 0; p < 4; ++p) {
      int rr = srow + p * 32;
      float4 av = *(const float4*)(A + (size_t)(m0 + rr) * lda + kc + scol);
      uint2 pa; pa.x = pk2(av.x, av.y); pa.y = pk2(av.z, av.w);
      *(uint2*)(&As[buf][rr * 40 + scol]) = pa;
      float4 bb = *(const float4*)(Bt + (size_t)(n0 + rr) * ldb + kc + scol);
      uint2 pb; pb.x = pk2(bb.x, bb.y); pb.y = pk2(bb.z, bb.w);
      *(uint2*)(&Bs[buf][rr * 40 + scol]) = pb;
    }
  };

  v8f acc[2][4] = {};
  stage(0, 0);
  int buf = 0;
  for (int kc = 0; kc < K; kc += 32, buf ^= 1) {
    __syncthreads();
    if (kc + 32 < K) stage(kc + 32, buf ^ 1);
    bv16 af[2], bfr[4];
    #pragma unroll
    for (int i = 0; i < 2; ++i)
      af[i] = lds_frag(&As[buf][(waveM * 32 + i * 16 + r16) * 40 + half * 8]);
    #pragma unroll
    for (int j = 0; j < 4; ++j)
      bfr[j] = lds_frag(&Bs[buf][(waveN * 64 + j * 16 + r16) * 40 + half * 8]);
    #pragma unroll
    for (int i = 0; i < 2; ++i)
      #pragma unroll
      for (int j = 0; j < 4; ++j)
        acc[i][j] = bmma(af[i], bfr[j], acc[i][j]);
  }

  const int halfM = half * 8;
  #pragma unroll
  for (int i = 0; i < 2; ++i)
    #pragma unroll
    for (int j = 0; j < 4; ++j) {
      int col = n0 + waveN * 64 + j * 16 + r16;
      #pragma unroll
      for (int rr = 0; rr < 8; ++rr) {
        int row = m0 + waveM * 32 + i * 16 + halfM + rr;
        float v = acc[i][j][rr] * alpha;
        if (bias) v += bias[col];
        if constexpr (POSENC) v += posenc_val(row & 2047, col);
        if constexpr (OUT_BF16)
          ((unsigned short*)Cptr)[(size_t)row * ldc + col] = f2bf(v);
        else
          ((float*)Cptr)[(size_t)row * ldc + col] = v;
      }
    }
}

// ---------------------------------------------------------------------------
// Flash attention, transposed-score form. Wave owns 16 queries (WMMA N dim,
// on lanes); keys on VGPR-index+half -> per-lane softmax stats and the P^T
// fragment is built directly in registers. Double-buffered K/V staging.
//   S^T = K_tile(16k x 64d) @ Q^T ; O^T(d x q) += V^T(d x 32k) @ P^T(32k x q)
// Q is pre-scaled by 1/sqrt(64).
// ---------------------------------------------------------------------------
__global__ __launch_bounds__(256) void flash_kernel(
    const unsigned short* __restrict__ Qb,
    const unsigned short* __restrict__ Kb,
    const unsigned short* __restrict__ Vb,
    float* __restrict__ O)
{
  const int S = 2048, E = 512, HD = 64;
  const int b = blockIdx.z, hh = blockIdx.y;
  const int t = threadIdx.x;
  const int wv = t >> 5, lane = t & 31, r16 = lane & 15, half = lane >> 4;
  const int q0 = blockIdx.x * 128 + wv * 16;
  const int jrow = t >> 3, cg = (t & 7) * 8;

  __shared__ __align__(16) unsigned short Ks[2][32 * 72];   // [key][d]
  __shared__ __align__(16) unsigned short Vts[2][64 * 40];  // [d][key]

  auto stage = [&](int kt, int buf) {
    const size_t base = (size_t)(b * S + kt + jrow) * E + hh * HD + cg;
    *(uint4*)(&Ks[buf][jrow * 72 + cg]) = *(const uint4*)(Kb + base);
    union { uint4 q; unsigned short s[8]; } vvu;
    vvu.q = *(const uint4*)(Vb + base);
    #pragma unroll
    for (int i = 0; i < 8; ++i) Vts[buf][(cg + i) * 40 + jrow] = vvu.s[i];
    if (kt + 32 < S) {  // prefetch chunk after next -> global_prefetch_b8
      __builtin_prefetch(Kb + base + (size_t)32 * E, 0, 1);
      __builtin_prefetch(Vb + base + (size_t)32 * E, 0, 1);
    }
  };

  bv16 bq[2];  // Q as B operand: lane n = query, k = head dim
  {
    const unsigned short* qrow = Qb + (size_t)(b * S + q0 + r16) * E + hh * HD;
    #pragma unroll
    for (int c = 0; c < 2; ++c)
      bq[c] = lds_frag(qrow + c * 32 + half * 8);
  }

  v8f o[4] = {};               // O^T: M = head dim (4 x 16), N = query
  float m = -1e30f, l = 0.f;   // per-lane (per-query-column) stats

  stage(0, 0);
  int buf = 0;
  for (int kt = 0; kt < S; kt += 32, buf ^= 1) {
    __syncthreads();
    if (kt + 32 < S) stage(kt + 32, buf ^ 1);

    // all fragments into distinct regs -> scheduler can overlap DS with VALU
    bv16 kf00 = lds_frag(&Ks[buf][r16 * 72 + half * 8]);
    bv16 kf01 = lds_frag(&Ks[buf][r16 * 72 + 32 + half * 8]);
    bv16 kf10 = lds_frag(&Ks[buf][(16 + r16) * 72 + half * 8]);
    bv16 kf11 = lds_frag(&Ks[buf][(16 + r16) * 72 + 32 + half * 8]);
    bv16 vf[4];
    #pragma unroll
    for (int nt = 0; nt < 4; ++nt)
      vf[nt] = lds_frag(&Vts[buf][(nt * 16 + r16) * 40 + half * 8]);

    v8f s0 = {}, s1 = {};  // S^T for keys 0-15 / 16-31 of this chunk
    s0 = bmma(kf00, bq[0], s0);
    s1 = bmma(kf10, bq[0], s1);
    s0 = bmma(kf01, bq[1], s0);
    s1 = bmma(kf11, bq[1], s1);

    // per-lane online softmax over the 32 keys of this chunk
    float mloc = -1e30f;
    #pragma unroll
    for (int r = 0; r < 8; ++r) mloc = fmaxf(mloc, fmaxf(s0[r], s1[r]));
    mloc = fmaxf(mloc, __shfl_xor(mloc, 16));
    float mn = fmaxf(m, mloc);
    float corr = __expf(m - mn);
    m = mn;

    union { bv16 v; unsigned int u[8]; } pf;  // P^T as B operand, in-register
    float ps = 0.f;
    #pragma unroll
    for (int r = 0; r < 8; r += 2) {
      float p0a = __expf(s0[r] - mn), p0b = __expf(s0[r + 1] - mn);
      float p1a = __expf(s1[r] - mn), p1b = __expf(s1[r + 1] - mn);
      ps += (p0a + p0b) + (p1a + p1b);
      pf.u[r >> 1]       = pk2(p0a, p0b);  // keys half*8+r    -> elems 0..7
      pf.u[4 + (r >> 1)] = pk2(p1a, p1b);  // keys 16+half*8+r -> elems 8..15
    }
    ps += __shfl_xor(ps, 16);
    l = l * corr + ps;

    #pragma unroll
    for (int nt = 0; nt < 4; ++nt)
      #pragma unroll
      for (int r = 0; r < 8; ++r) o[nt][r] *= corr;

    #pragma unroll
    for (int nt = 0; nt < 4; ++nt)
      o[nt] = bmma(vf[nt], pf.v, o[nt]);
  }

  const float inv = 1.f / l;
  float* orow = O + (size_t)(b * S + q0 + r16) * E + hh * HD + half * 8;
  #pragma unroll
  for (int nt = 0; nt < 4; ++nt) {
    float4 w0, w1;
    w0.x = o[nt][0] * inv; w0.y = o[nt][1] * inv;
    w0.z = o[nt][2] * inv; w0.w = o[nt][3] * inv;
    w1.x = o[nt][4] * inv; w1.y = o[nt][5] * inv;
    w1.z = o[nt][6] * inv; w1.w = o[nt][7] * inv;
    *(float4*)(orow + nt * 16)     = w0;
    *(float4*)(orow + nt * 16 + 4) = w1;
  }
}

// ---------------------------------------------------------------------------
// Fused FFN: out = relu(cos(h[:, :8]+phi) @ W1^T) @ W2^T   (K=2048)
// A-tile produced on the fly; double-buffered, produceA(next) overlaps WMMA.
// ---------------------------------------------------------------------------
__global__ __launch_bounds__(256) void ffn_kernel(
    const float* __restrict__ h, const float* __restrict__ phi,
    const float* __restrict__ W1, const float* __restrict__ W2,
    float* __restrict__ out)
{
  __shared__ float qf[128][8];
  __shared__ float w1c[2][32][8];
  __shared__ __align__(16) unsigned short As[2][128 * 40];
  __shared__ __align__(16) unsigned short Bs[2][128 * 40];

  const int t = threadIdx.x;
  const int m0 = blockIdx.y * 128, n0 = blockIdx.x * 128;
  const int lane = t & 31, r16 = lane & 15, half = lane >> 4;
  const int wv = t >> 5, waveM = wv >> 1, waveN = wv & 1;
  const int srow = t >> 3, scol = (t & 7) * 4;

  auto stageW = [&](int kc, int buf) {
    w1c[buf][t >> 3][t & 7] = W1[(size_t)(kc + (t >> 3)) * 8 + (t & 7)];
    #pragma unroll
    for (int p = 0; p < 4; ++p) {
      int rr = srow + p * 32;
      float4 bb = *(const float4*)(W2 + (size_t)(n0 + rr) * 2048 + kc + scol);
      uint2 pb; pb.x = pk2(bb.x, bb.y); pb.y = pk2(bb.z, bb.w);
      *(uint2*)(&Bs[buf][rr * 40 + scol]) = pb;
    }
  };
  auto produceA = [&](int buf) {
    #pragma unroll
    for (int p = 0; p < 4; ++p) {
      int row = srow + p * 32;
      float vals[4];
      #pragma unroll
      for (int cc = 0; cc < 4; ++cc) {
        float s = 0.f;
        #pragma unroll
        for (int w = 0; w < 8; ++w) s += qf[row][w] * w1c[buf][scol + cc][w];
        vals[cc] = fmaxf(s, 0.f);
      }
      uint2 pw; pw.x = pk2(vals[0], vals[1]); pw.y = pk2(vals[2], vals[3]);
      *(uint2*)(&As[buf][row * 40 + scol]) = pw;
    }
  };

  {
    int idx = t * 4;
    #pragma unroll
    for (int k = 0; k < 4; ++k) {
      int row = (idx + k) >> 3, w = (idx + k) & 7;
      qf[row][w] = __cosf(h[(size_t)(m0 + row) * 512 + w] + phi[w]);
    }
  }
  stageW(0, 0);
  __syncthreads();
  produceA(0);

  v8f acc[2][4] = {};
  int buf = 0;
  for (int kc = 0; kc < 2048; kc += 32, buf ^= 1) {
    __syncthreads();                       // As/Bs(buf) ready; buf^1 reusable
    if (kc + 32 < 2048) stageW(kc + 32, buf ^ 1);
    bv16 af[2], bfr[4];
    #pragma unroll
    for (int i = 0; i < 2; ++i)
      af[i] = lds_frag(&As[buf][(waveM * 32 + i * 16 + r16) * 40 + half * 8]);
    #pragma unroll
    for (int j = 0; j < 4; ++j)
      bfr[j] = lds_frag(&Bs[buf][(waveN * 64 + j * 16 + r16) * 40 + half * 8]);
    __syncthreads();                       // w1c(buf^1) staged by all waves
    if (kc + 32 < 2048) produceA(buf ^ 1);
    #pragma unroll
    for (int i = 0; i < 2; ++i)
      #pragma unroll
      for (int j = 0; j < 4; ++j)
        acc[i][j] = bmma(af[i], bfr[j], acc[i][j]);
  }

  const int halfM = half * 8;
  #pragma unroll
  for (int i = 0; i < 2; ++i)
    #pragma unroll
    for (int j = 0; j < 4; ++j) {
      int col = n0 + waveN * 64 + j * 16 + r16;
      #pragma unroll
      for (int rr = 0; rr < 8; ++rr) {
        int row = m0 + waveM * 32 + i * 16 + halfM + rr;
        out[(size_t)row * 512 + col] = acc[i][j][rr];
      }
    }
}

// ---------------------------------------------------------------------------
// Weff[f][w] = sum_e W[f][e] * Wproj[e][w]  for W in {Wq,Wk,Wv}.
// (QKV projections factor through the rank-8 quantum layer.)
// ---------------------------------------------------------------------------
__global__ __launch_bounds__(256) void weff_kernel(
    const float* __restrict__ Wq, const float* __restrict__ Wk,
    const float* __restrict__ Wv, const float* __restrict__ Wproj,
    float* __restrict__ weff)
{
  const float* W = (blockIdx.y == 0) ? Wq : (blockIdx.y == 1) ? Wk : Wv;
  int idx = blockIdx.x * 256 + threadIdx.x;  // 0..4095
  int f = idx >> 3, w = idx & 7;
  float s = 0.f;
  for (int e = 0; e < 512; ++e) s += W[(size_t)f * 512 + e] * Wproj[e * 8 + w];
  weff[(size_t)blockIdx.y * 4096 + idx] = s;
}

// ---------------------------------------------------------------------------
// Q/K/V[row,e] = dot8(cos(h[row,:8]+theta), Weff_{q,k,v}[e,:])  -> bf16
// Q additionally scaled by 1/sqrt(64). One block per token row.
// ---------------------------------------------------------------------------
__global__ __launch_bounds__(256) void qkv_kernel(
    const float* __restrict__ h, const float* __restrict__ theta,
    const float* __restrict__ weff,
    unsigned short* __restrict__ Qb, unsigned short* __restrict__ Kb,
    unsigned short* __restrict__ Vb)
{
  __shared__ float cw[8];
  const int row = blockIdx.x;
  if (threadIdx.x < 8)
    cw[threadIdx.x] = __cosf(h[(size_t)row * 512 + threadIdx.x] + theta[threadIdx.x]);
  __syncthreads();
  #pragma unroll
  for (int pass = 0; pass < 2; ++pass) {
    int e = threadIdx.x + pass * 256;
    const float* wq = weff + (size_t)e * 8;
    const float* wk = weff + 4096 + (size_t)e * 8;
    const float* wvv = weff + 8192 + (size_t)e * 8;
    float sq = 0.f, sk = 0.f, sv = 0.f;
    #pragma unroll
    for (int w = 0; w < 8; ++w) {
      float c = cw[w];
      sq += c * wq[w]; sk += c * wk[w]; sv += c * wvv[w];
    }
    Qb[(size_t)row * 512 + e] = f2bf(sq * 0.125f);
    Kb[(size_t)row * 512 + e] = f2bf(sk);
    Vb[(size_t)row * 512 + e] = f2bf(sv);
  }
}

// h = LayerNorm(h + addin) * g + b     (one block per 512-wide row)
__global__ __launch_bounds__(256) void ln_kernel(
    float* __restrict__ h, const float* __restrict__ addin,
    const float* __restrict__ g, const float* __restrict__ bta)
{
  const int row = blockIdx.x, t = threadIdx.x;
  const int lane = t & 31, wv = t >> 5;
  float v0 = h[(size_t)row * 512 + t]       + addin[(size_t)row * 512 + t];
  float v1 = h[(size_t)row * 512 + t + 256] + addin[(size_t)row * 512 + t + 256];
  float s1 = v0 + v1, s2 = v0 * v0 + v1 * v1;
  #pragma unroll
  for (int off = 16; off > 0; off >>= 1) {
    s1 += __shfl_xor(s1, off);
    s2 += __shfl_xor(s2, off);
  }
  __shared__ float sh1[8], sh2[8];
  if (lane == 0) { sh1[wv] = s1; sh2[wv] = s2; }
  __syncthreads();
  float t1 = 0.f, t2 = 0.f;
  #pragma unroll
  for (int i = 0; i < 8; ++i) { t1 += sh1[i]; t2 += sh2[i]; }
  float mean = t1 * (1.f / 512.f);
  float var  = t2 * (1.f / 512.f) - mean * mean;
  float rs = rsqrtf(var + 1e-5f);
  h[(size_t)row * 512 + t]       = (v0 - mean) * rs * g[t]       + bta[t];
  h[(size_t)row * 512 + t + 256] = (v1 - mean) * rs * g[t + 256] + bta[t + 256];
}

// Two-stage mean pool: stage 1 reduces 128 seq rows per block (8x16 blocks).
__global__ __launch_bounds__(512) void pool1_kernel(const float* __restrict__ h,
                                                    float* __restrict__ partial)
{
  const int b = blockIdx.x, chunk = blockIdx.y, e = threadIdx.x;
  const float* base = h + (size_t)(b * 2048 + chunk * 128) * 512 + e;
  float s = 0.f;
  for (int i = 0; i < 128; ++i) s += base[(size_t)i * 512];
  partial[(size_t)(b * 16 + chunk) * 512 + e] = s;
}

__global__ __launch_bounds__(512) void pool2_kernel(const float* __restrict__ partial,
                                                    float* __restrict__ pooled)
{
  const int b = blockIdx.x, e = threadIdx.x;
  float s = 0.f;
  #pragma unroll
  for (int c = 0; c < 16; ++c) s += partial[(size_t)(b * 16 + c) * 512 + e];
  pooled[b * 512 + e] = s * (1.f / 2048.f);
}

// Classifier: one wave per class; lanes split K (coalesced Wc reads), shuffle
// reduce, lane 0 writes all 8 batches. 1000 classes = 125 blocks x 8 waves.
__global__ __launch_bounds__(256) void cls_kernel(const float* __restrict__ pooled,
                                                  const float* __restrict__ Wc,
                                                  const float* __restrict__ bc,
                                                  float* __restrict__ out)
{
  const int wv = threadIdx.x >> 5, lane = threadIdx.x & 31;
  const int c = blockIdx.x * 8 + wv;
  const float* w = Wc + (size_t)c * 512;
  float acc[8] = {};
  for (int k = lane; k < 512; k += 32) {
    float wk = w[k];
    #pragma unroll
    for (int b = 0; b < 8; ++b) acc[b] += wk * pooled[b * 512 + k];
  }
  #pragma unroll
  for (int b = 0; b < 8; ++b) {
    #pragma unroll
    for (int off = 16; off > 0; off >>= 1) acc[b] += __shfl_xor(acc[b], off);
  }
  if (lane == 0) {
    #pragma unroll
    for (int b = 0; b < 8; ++b) out[b * 1000 + c] = acc[b] + bc[c];
  }
}

// ---------------------------------------------------------------------------
extern "C" void kernel_launch(void* const* d_in, const int* in_sizes, int n_in,
                              void* d_out, int out_size, void* d_ws, size_t ws_size,
                              hipStream_t stream)
{
  (void)in_sizes; (void)n_in; (void)out_size; (void)ws_size;
  const float* x     = (const float*)d_in[0];
  const float* Wi    = (const float*)d_in[1];
  const float* bi    = (const float*)d_in[2];
  const float* theta = (const float*)d_in[3];
  const float* Wproj = (const float*)d_in[4];
  const float* Wq    = (const float*)d_in[5];
  const float* Wk    = (const float*)d_in[6];
  const float* Wv    = (const float*)d_in[7];
  const float* Wo    = (const float*)d_in[8];
  const float* g1    = (const float*)d_in[9];
  const float* b1    = (const float*)d_in[10];
  const float* phi   = (const float*)d_in[11];
  const float* W1    = (const float*)d_in[12];
  const float* W2    = (const float*)d_in[13];
  const float* g2    = (const float*)d_in[14];
  const float* b2    = (const float*)d_in[15];
  const float* Wc    = (const float*)d_in[16];
  const float* bc    = (const float*)d_in[17];
  float* out = (float*)d_out;

  char* ws = (char*)d_ws;
  float*          h       = (float*)(ws);                                // 32 MB
  float*          tA      = (float*)(ws + ((size_t)32  << 20));          // 32 MB
  float*          tB      = (float*)(ws + ((size_t)64  << 20));          // 32 MB
  unsigned short* Qb      = (unsigned short*)(ws + ((size_t)96  << 20)); // 16 MB
  unsigned short* Kb      = (unsigned short*)(ws + ((size_t)112 << 20)); // 16 MB
  unsigned short* Vb      = (unsigned short*)(ws + ((size_t)128 << 20)); // 16 MB
  float*          pooled  = (float*)(ws + ((size_t)144 << 20));          // 16 KB
  float*          weff    = (float*)(ws + ((size_t)145 << 20));          // 48 KB
  float*          partial = (float*)(ws + ((size_t)146 << 20));          // 256 KB

  dim3 blk(256);
  dim3 gg(4, 128);  // N/128=4, M/128=128  (M=16384, N=512)

  // h = x @ Wi^T + bi + posenc
  gemm_kernel<false, true><<<gg, blk, 0, stream>>>(x, 256, Wi, 256, h, 512, 256,
                                                   bi, 1.0f);

  for (int l = 0; l < 2; ++l) {
    const float* wq  = Wq + (size_t)l * 512 * 512;
    const float* wk  = Wk + (size_t)l * 512 * 512;
    const float* wvv = Wv + (size_t)l * 512 * 512;
    // Weff_{q,k,v} = W @ Wproj  (rank-8 factorization of QKV projections)
    weff_kernel<<<dim3(16, 3), blk, 0, stream>>>(wq, wk, wvv,
                                                 Wproj + (size_t)l * 512 * 8, weff);
    qkv_kernel<<<16384, blk, 0, stream>>>(h, theta + l * 8, weff, Qb, Kb, Vb);

    flash_kernel<<<dim3(16, 8, 8), blk, 0, stream>>>(Qb, Kb, Vb, tA);

    gemm_kernel<false, false><<<gg, blk, 0, stream>>>(tA, 512,
        Wo + (size_t)l * 512 * 512, 512, tB, 512, 512, nullptr, 1.0f);
    ln_kernel<<<16384, blk, 0, stream>>>(h, tB, g1 + l * 512, b1 + l * 512);

    ffn_kernel<<<gg, blk, 0, stream>>>(h, phi + l * 8, W1 + (size_t)l * 2048 * 8,
                                       W2 + (size_t)l * 512 * 2048, tB);
    ln_kernel<<<16384, blk, 0, stream>>>(h, tB, g2 + l * 512, b2 + l * 512);
  }

  pool1_kernel<<<dim3(8, 16), 512, 0, stream>>>(h, partial);
  pool2_kernel<<<8, 512, 0, stream>>>(partial, pooled);
  cls_kernel<<<125, blk, 0, stream>>>(pooled, Wc, bc, out);
}
